// CrossAttention_56143812493519
// MI455X (gfx1250) — compile-verified, matrix operand order checked by reading, not run on hardware
//
#include <hip/hip_runtime.h>

// ---------------------------------------------------------------------------
// CrossAttention for MI455X (gfx1250):
//   - all GEMMs via v_wmma_f32_16x16x32_bf16 (f32 accumulate)
//   - double-buffered LDS staging, BK=64 (16 WMMA per barrier)
//   - global->LDS staging through GLOBAL_LOAD_ASYNC_TO_LDS_B128
//   - fragments preloaded per K-step so DScnt waits stay partial
// ---------------------------------------------------------------------------

typedef __bf16 bf16;
typedef __attribute__((ext_vector_type(16))) __bf16 v16bf;
typedef __attribute__((ext_vector_type(8)))  __bf16 v8bf;
typedef __attribute__((ext_vector_type(8)))  float  v8f;
typedef __attribute__((ext_vector_type(4)))  float  v4f;
typedef __attribute__((ext_vector_type(4)))  int    v4i;

constexpr int kB  = 16;
constexpr int kNi = 2048;
constexpr int kNt = 1024;
constexpr int kD  = 1024;

#if defined(__has_builtin)
#  if __has_builtin(__builtin_amdgcn_global_load_async_to_lds_b128)
#    define HAVE_ASYNC_LDS 1
#  endif
#endif
#ifndef HAVE_ASYNC_LDS
#  define HAVE_ASYNC_LDS 0
#endif

// pointee types per the builtin's actual signature: <4 x i32> in AS1 / AS3
typedef __attribute__((address_space(1))) v4i gv4i;
typedef __attribute__((address_space(3))) v4i lv4i;

__device__ __forceinline__ void async_copy_b128(const bf16* gsrc, bf16* ldst)
{
#if HAVE_ASYNC_LDS
    __builtin_amdgcn_global_load_async_to_lds_b128(
        (gv4i*)gsrc, (lv4i*)ldst, /*offset=*/0, /*cpol=*/0);
#else
    *(v8bf*)ldst = *(const v8bf*)gsrc;
#endif
}

__device__ __forceinline__ void async_wait_all()
{
#if HAVE_ASYNC_LDS
#  if __has_builtin(__builtin_amdgcn_s_wait_asynccnt)
    __builtin_amdgcn_s_wait_asynccnt(0);
#  else
    asm volatile("s_wait_asynccnt 0" ::: "memory");
#  endif
#endif
}

// ---------------------------------------------------------------------------
// LayerNorm (one block per row of 1024 f32), writes bf16
// ---------------------------------------------------------------------------
__global__ void __launch_bounds__(256)
ln_bf16_kernel(const float* __restrict__ x, const float* __restrict__ g,
               const float* __restrict__ b, bf16* __restrict__ out)
{
    __shared__ float red[256];
    const int t = threadIdx.x;
    const size_t row = blockIdx.x;
    const float* xr = x + row * (size_t)kD;
    bf16* yr = out + row * (size_t)kD;

    v4f v = *(const v4f*)(xr + t * 4);
    red[t] = v.x + v.y + v.z + v.w;
    __syncthreads();
    for (int s = 128; s > 0; s >>= 1) {
        if (t < s) red[t] += red[t + s];
        __syncthreads();
    }
    const float mu = red[0] * (1.0f / kD);
    __syncthreads();

    float d0 = v.x - mu, d1 = v.y - mu, d2 = v.z - mu, d3 = v.w - mu;
    red[t] = d0 * d0 + d1 * d1 + d2 * d2 + d3 * d3;
    __syncthreads();
    for (int s = 128; s > 0; s >>= 1) {
        if (t < s) red[t] += red[t + s];
        __syncthreads();
    }
    const float rstd = rsqrtf(red[0] * (1.0f / kD) + 1e-5f);

    v4f gv = *(const v4f*)(g + t * 4);
    v4f bv = *(const v4f*)(b + t * 4);
    yr[t * 4 + 0] = (bf16)(d0 * rstd * gv.x + bv.x);
    yr[t * 4 + 1] = (bf16)(d1 * rstd * gv.y + bv.y);
    yr[t * 4 + 2] = (bf16)(d2 * rstd * gv.z + bv.z);
    yr[t * 4 + 3] = (bf16)(d3 * rstd * gv.w + bv.w);
}

// ---------------------------------------------------------------------------
// f32 [R x C] -> bf16 transposed [C x R]  (32x32 LDS tile transpose)
// ---------------------------------------------------------------------------
__global__ void __launch_bounds__(256)
cvt_bf16_transpose_kernel(const float* __restrict__ in, bf16* __restrict__ out,
                          int R, int C)
{
    __shared__ float tile[32][33];
    const int r0 = blockIdx.y * 32, c0 = blockIdx.x * 32;
    const int tr = threadIdx.x & 31;        // fast dim
    const int tc = threadIdx.x >> 5;        // 0..7
#pragma unroll
    for (int i = 0; i < 32; i += 8)
        tile[tc + i][tr] = in[(size_t)(r0 + tc + i) * C + (c0 + tr)];
    __syncthreads();
#pragma unroll
    for (int i = 0; i < 32; i += 8)
        out[(size_t)(c0 + tc + i) * R + (r0 + tr)] = (bf16)tile[tr][tc + i];
}

// ---------------------------------------------------------------------------
// Row softmax over 1024 f32 columns, writes bf16 probabilities
// ---------------------------------------------------------------------------
__global__ void __launch_bounds__(256)
softmax_kernel(const float* __restrict__ S, bf16* __restrict__ P)
{
    __shared__ float red[256];
    const int t = threadIdx.x;
    const size_t row = blockIdx.x;
    const float* sr = S + row * (size_t)kNt;
    bf16* pr = P + row * (size_t)kNt;

    v4f v = *(const v4f*)(sr + t * 4);
    red[t] = fmaxf(fmaxf(v.x, v.y), fmaxf(v.z, v.w));
    __syncthreads();
    for (int s = 128; s > 0; s >>= 1) {
        if (t < s) red[t] = fmaxf(red[t], red[t + s]);
        __syncthreads();
    }
    const float rowmax = red[0];
    __syncthreads();

    float e0 = __expf(v.x - rowmax), e1 = __expf(v.y - rowmax);
    float e2 = __expf(v.z - rowmax), e3 = __expf(v.w - rowmax);
    red[t] = e0 + e1 + e2 + e3;
    __syncthreads();
    for (int s = 128; s > 0; s >>= 1) {
        if (t < s) red[t] += red[t + s];
        __syncthreads();
    }
    const float inv = 1.0f / red[0];
    pr[t * 4 + 0] = (bf16)(e0 * inv);
    pr[t * 4 + 1] = (bf16)(e1 * inv);
    pr[t * 4 + 2] = (bf16)(e2 * inv);
    pr[t * 4 + 3] = (bf16)(e3 * inv);
}

// ---------------------------------------------------------------------------
// Batched WMMA GEMM: C[b] = op(A[b]) @ op(B[b]) * scale (+ bias)
//   TA=false: A physical [M x K].  TA=true: A physical [K x M] (A = phys^T).
//   TB=false: B physical [K x N].  TB=true: B physical [N x K] (B = phys^T).
// Block tile 128x128, K-step 64, double-buffered LDS, 8 waves (4x2 grid of
// 32x64 wave tiles). A-tile LDS [m][k], B-tile LDS [n][k]: WMMA fragments are
// two contiguous 16B LDS reads per lane matching the 16-bit A/B VGPR layout.
// Requires M,N % 128 == 0 and K % 64 == 0 (true for all calls here).
// ---------------------------------------------------------------------------
template <bool TA, bool TB, bool OUT_BF16>
__global__ void __launch_bounds__(256)
gemm_wmma_kernel(const bf16* __restrict__ A, long strideA,
                 const bf16* __restrict__ B, long strideB,
                 void* __restrict__ Cv, long strideC,
                 const float* __restrict__ bias, float scale,
                 int M, int N, int K)
{
    constexpr int BM = 128, BN = 128, BK = 64;
    constexpr int LD = BK + 8;                    // 72 bf16 = 144 B (16B aligned)
    __shared__ alignas(16) bf16 sA[2][BM * LD];   // [stage][m][k]
    __shared__ alignas(16) bf16 sB[2][BN * LD];   // [stage][n][k]

    const int tid  = threadIdx.x;
    const int lane = tid & 31;
    const int wave = tid >> 5;
    const int wm = (wave & 3) * 32;               // wave M offset inside tile
    const int wn = (wave >> 2) * 64;              // wave N offset inside tile
    const int tileM = blockIdx.y * BM;
    const int tileN = blockIdx.x * BN;

    const bf16* Ab = A + (long)blockIdx.z * strideA;
    const bf16* Bb = B + (long)blockIdx.z * strideB;

    v8f acc[2][4];
#pragma unroll
    for (int m = 0; m < 2; ++m)
#pragma unroll
        for (int n = 0; n < 4; ++n)
            acc[m][n] = (v8f){0.f, 0.f, 0.f, 0.f, 0.f, 0.f, 0.f, 0.f};

    const int mrow  = lane & 15;                  // fragment row/col index
    const int halfk = (lane >> 4) * 8;            // per-half K sub-offset

    // ---- tile stagers (each thread moves 32 bf16 per tile) ----
    auto stageA = [&](int k0, int st) {
        if (!TA) {
            const int r = tid >> 1, c = (tid & 1) * 32;
            const bf16* src = Ab + (long)(tileM + r) * K + k0 + c;
            bf16* dst = &sA[st][r * LD + c];
#pragma unroll
            for (int j = 0; j < 4; ++j)
                async_copy_b128(src + 8 * j, dst + 8 * j);
        } else {
            // physical [K x M]: gather columns into rows of sA (2B scatter)
            const int kk = tid >> 2;              // 0..63
            const int m0 = (tid & 3) * 32;        // 0..96
            const bf16* src = Ab + (long)(k0 + kk) * M + tileM + m0;
#pragma unroll
            for (int j4 = 0; j4 < 4; ++j4) {
                v8bf v = *(const v8bf*)(src + 8 * j4);
#pragma unroll
                for (int j = 0; j < 8; ++j)
                    sA[st][(m0 + 8 * j4 + j) * LD + kk] = v[j];
            }
        }
    };
    auto stageB = [&](int k0, int st) {
        if (TB) {
            const int r = tid >> 1, c = (tid & 1) * 32;
            const bf16* src = Bb + (long)(tileN + r) * K + k0 + c;
            bf16* dst = &sB[st][r * LD + c];
#pragma unroll
            for (int j = 0; j < 4; ++j)
                async_copy_b128(src + 8 * j, dst + 8 * j);
        } else {
            // physical [K x N]: transpose scatter into [n][k]
            const int kk = tid >> 2;
            const int n0 = (tid & 3) * 32;
            const bf16* src = Bb + (long)(k0 + kk) * N + tileN + n0;
#pragma unroll
            for (int j4 = 0; j4 < 4; ++j4) {
                v8bf v = *(const v8bf*)(src + 8 * j4);
#pragma unroll
                for (int j = 0; j < 8; ++j)
                    sB[st][(n0 + 8 * j4 + j) * LD + kk] = v[j];
            }
        }
    };

    // ---- software pipeline: prefetch stage s^1 while computing stage s ----
    stageA(0, 0);
    stageB(0, 0);
    async_wait_all();
    __syncthreads();

    int s = 0;
    for (int k0 = 0; k0 < K; k0 += BK, s ^= 1) {
        if (k0 + BK < K) {
            stageA(k0 + BK, s ^ 1);
            stageB(k0 + BK, s ^ 1);
        }
        // 2 K-steps of 32 -> 16 WMMAs on stage s; all fragments of a K-step
        // are loaded before the first WMMA so DScnt waits can stay partial.
#pragma unroll
        for (int ks = 0; ks < BK; ks += 32) {
            v16bf afrag[2], bfrag[4];
#pragma unroll
            for (int m = 0; m < 2; ++m) {
                const bf16* p = &sA[s][(wm + m * 16 + mrow) * LD + ks + halfk];
                v8bf lo = *(const v8bf*)p;          // K = ks + halfk + 0..7
                v8bf hi = *(const v8bf*)(p + 16);   // K = ks + 16 + halfk + 0..7
                afrag[m] = __builtin_shufflevector(lo, hi, 0, 1, 2, 3, 4, 5, 6, 7,
                                                   8, 9, 10, 11, 12, 13, 14, 15);
            }
#pragma unroll
            for (int n = 0; n < 4; ++n) {
                const bf16* p = &sB[s][(wn + n * 16 + mrow) * LD + ks + halfk];
                v8bf lo = *(const v8bf*)p;
                v8bf hi = *(const v8bf*)(p + 16);
                bfrag[n] = __builtin_shufflevector(lo, hi, 0, 1, 2, 3, 4, 5, 6, 7,
                                                   8, 9, 10, 11, 12, 13, 14, 15);
            }
#pragma unroll
            for (int n = 0; n < 4; ++n)
#pragma unroll
                for (int m = 0; m < 2; ++m)
                    acc[m][n] = __builtin_amdgcn_wmma_f32_16x16x32_bf16(
                        /*neg_a=*/false, afrag[m], /*neg_b=*/false, bfrag[n],
                        /*c_mod=*/(short)0, acc[m][n],
                        /*reuse_a=*/false, /*reuse_b=*/false);
        }
        async_wait_all();
        __syncthreads();
    }

    // ---- epilogue: VGPR r -> M = 16m + 8*(lane/16) + r, N = 16n + lane%16 ----
    const int rowHalf = (lane >> 4) * 8;
    const long cBase = (long)blockIdx.z * strideC;
#pragma unroll
    for (int m = 0; m < 2; ++m) {
#pragma unroll
        for (int n = 0; n < 4; ++n) {
            const int gn = tileN + wn + n * 16 + mrow;
            const float bv = bias ? bias[gn] : 0.0f;
#pragma unroll
            for (int r = 0; r < 8; ++r) {
                const int gm = tileM + wm + m * 16 + rowHalf + r;
                const float val = acc[m][n][r] * scale + bv;
                if (OUT_BF16)
                    ((bf16*)Cv)[cBase + (long)gm * N + gn] = (bf16)val;
                else
                    ((float*)Cv)[cBase + (long)gm * N + gn] = val;
            }
        }
    }
}

// ---------------------------------------------------------------------------
// Host-side orchestration
// ---------------------------------------------------------------------------
extern "C" void kernel_launch(void* const* d_in, const int* in_sizes, int n_in,
                              void* d_out, int out_size, void* d_ws, size_t ws_size,
                              hipStream_t stream)
{
    const float* img  = (const float*)d_in[0];
    const float* txt  = (const float*)d_in[1];
    const float* ln_g = (const float*)d_in[2];
    const float* ln_b = (const float*)d_in[3];
    const float* Wq   = (const float*)d_in[4];
    const float* bq   = (const float*)d_in[5];
    const float* Wk   = (const float*)d_in[6];
    const float* bk   = (const float*)d_in[7];
    const float* Wv   = (const float*)d_in[8];
    const float* bv   = (const float*)d_in[9];

    float* image_out = (float*)d_out;                      // [B,Ni,D]
    float* text_out  = image_out + (size_t)kB * kNi * kD;  // [B,Nt,D]

    char* ws = (char*)d_ws;
    size_t off = 0;
    auto alloc = [&](size_t bytes) -> void* {
        void* p = ws + off;
        off += (bytes + 255) & ~(size_t)255;
        return p;
    };
    bf16* imgn  = (bf16*)alloc((size_t)kB * kNi * kD * sizeof(bf16));
    bf16* txtn  = (bf16*)alloc((size_t)kB * kNt * kD * sizeof(bf16));
    bf16* WqT   = (bf16*)alloc((size_t)kD * kD * sizeof(bf16));   // [N x K] = Wq^T
    bf16* WkT   = (bf16*)alloc((size_t)kD * kD * sizeof(bf16));
    bf16* WvT   = (bf16*)alloc((size_t)kD * kD * sizeof(bf16));
    bf16* Qbf   = (bf16*)alloc((size_t)kB * kNi * kD * sizeof(bf16));
    bf16* Kbf   = (bf16*)alloc((size_t)kB * kNt * kD * sizeof(bf16));
    bf16* Vbf   = (bf16*)alloc((size_t)kB * kNt * kD * sizeof(bf16));
    bf16* VIbf  = (bf16*)alloc((size_t)kB * kNi * kD * sizeof(bf16));
    float* Sf32 = (float*)alloc((size_t)kB * kNi * kNt * sizeof(float));
    bf16* Pbf   = (bf16*)alloc((size_t)kB * kNi * kNt * sizeof(bf16));

    const float attnScale = 1.0f / 32.0f;   // D^-0.5, D = 1024

    // 1) LayerNorm -> bf16 activations
    ln_bf16_kernel<<<kB * kNi, 256, 0, stream>>>(img, ln_g, ln_b, imgn);
    ln_bf16_kernel<<<kB * kNt, 256, 0, stream>>>(txt, ln_g, ln_b, txtn);

    // 2) Weights f32 [K x N] -> bf16 transposed [N x K] (enables TB fast path)
    cvt_bf16_transpose_kernel<<<dim3(kD / 32, kD / 32), 256, 0, stream>>>(Wq, WqT, kD, kD);
    cvt_bf16_transpose_kernel<<<dim3(kD / 32, kD / 32), 256, 0, stream>>>(Wk, WkT, kD, kD);
    cvt_bf16_transpose_kernel<<<dim3(kD / 32, kD / 32), 256, 0, stream>>>(Wv, WvT, kD, kD);

    // 3) Projections (bf16 out, f32 accumulate, + bias), all contiguous TB path
    gemm_wmma_kernel<false, true, true><<<dim3(kD / 128, kNi / 128, kB), 256, 0, stream>>>(
        imgn, (long)kNi * kD, WqT, 0, Qbf, (long)kNi * kD, bq, 1.0f, kNi, kD, kD);
    gemm_wmma_kernel<false, true, true><<<dim3(kD / 128, kNt / 128, kB), 256, 0, stream>>>(
        txtn, (long)kNt * kD, WkT, 0, Kbf, (long)kNt * kD, bk, 1.0f, kNt, kD, kD);
    gemm_wmma_kernel<false, true, true><<<dim3(kD / 128, kNt / 128, kB), 256, 0, stream>>>(
        txtn, (long)kNt * kD, WvT, 0, Vbf, (long)kNt * kD, bv, 1.0f, kNt, kD, kD);
    gemm_wmma_kernel<false, true, true><<<dim3(kD / 128, kNi / 128, kB), 256, 0, stream>>>(
        imgn, (long)kNi * kD, WvT, 0, VIbf, (long)kNi * kD, bv, 1.0f, kNi, kD, kD);

    // 4) S = scale * Q @ K^T   (B physical is K [Nt x D] -> TB)
    gemm_wmma_kernel<false, true, false><<<dim3(kNt / 128, kNi / 128, kB), 256, 0, stream>>>(
        Qbf, (long)kNi * kD, Kbf, (long)kNt * kD, Sf32, (long)kNi * kNt,
        nullptr, attnScale, kNi, kNt, kD);

    // 5) P = softmax_rows(S) -> bf16
    softmax_kernel<<<kB * kNi, 256, 0, stream>>>(Sf32, Pbf);

    // 6) image_out = P @ V
    gemm_wmma_kernel<false, false, false><<<dim3(kD / 128, kNi / 128, kB), 256, 0, stream>>>(
        Pbf, (long)kNi * kNt, Vbf, (long)kNt * kD, image_out, (long)kNi * kD,
        nullptr, 1.0f, kNi, kD, kNt);

    // 7) text_out = P^T @ V_img   (A physical is P [Ni x Nt] -> TA)
    gemm_wmma_kernel<true, false, false><<<dim3(kD / 128, kNt / 128, kB), 256, 0, stream>>>(
        Pbf, (long)kNi * kNt, VIbf, (long)kNi * kD, text_out, (long)kNt * kD,
        nullptr, 1.0f, kNt, kD, kNi);
}
